// MultiHeadAttModel_21973052686431
// MI455X (gfx1250) — matrix operand order repaired and checked by reading, not compiled, not run on hardware
//
#include <hip/hip_runtime.h>

// ---------------------------------------------------------------------------
// MultiHeadAttModel on MI455X (gfx1250): bf16 WMMA GEMM conv pipeline.
//  - conv layers as WMMA GEMMs (bf16 in, fp32 accumulate)
//  - layer-2 K-dim permuted tau-major; activations stored time-major so the
//    im2col B-tile staging is pure global_load_b128 (no scalar gather)
//  - conv2-key epilogue fuses the q.k score dot (k never materialized)
//  - value/DCT/linW algebra fused into a tiny per-batch finishing kernel
// ---------------------------------------------------------------------------

typedef __attribute__((ext_vector_type(16))) __bf16 v16bf;
typedef __attribute__((ext_vector_type(8)))  float  v8f;

union Frag { uint4 u[2]; v16bf v; };

// problem constants
constexpr int cH  = 4;
constexpr int cD  = 512;
constexpr int cF  = 33;
constexpr int cBS = 2048;
constexpr int cCB = 256;          // batch chunk processed per GEMM launch
constexpr int cNCHUNK = cBS / cCB;

// K dims (padded to multiples of 32 for WMMA k-steps)
constexpr int K1_REAL = cF * 6;   // 198
constexpr int K1_PAD  = 224;      // 7 wmma k-steps
constexpr int K2      = cD * 5;   // 2560 = 80 k-steps (tau-major: k = tau*512 + i)

// workspace layout (bytes)
constexpr size_t SZ_W1  = (size_t)cH * cD * K1_PAD * 2;   // bf16
constexpr size_t SZ_W2  = (size_t)cH * cD * K2 * 2;       // bf16
constexpr size_t SZ_BN1 = (size_t)cH * cD * 4;            // fp32 per array
constexpr size_t OFF_WK1 = 0;
constexpr size_t OFF_WQ1 = OFF_WK1 + SZ_W1;
constexpr size_t OFF_WK2 = OFF_WQ1 + SZ_W1;
constexpr size_t OFF_WQ2 = OFF_WK2 + SZ_W2;
constexpr size_t OFF_BN  = OFF_WQ2 + SZ_W2;               // 8 arrays: sk1,tk1,sk2,tk2,sq1,tq1,sq2,tq2
constexpr size_t OFF_M   = OFF_BN + 8 * SZ_BN1;           // 20*4*35 fp32
constexpr size_t OFF_X1K = OFF_M + 11264;                 // bf16 [h][b][t=20][d]
constexpr size_t SZ_X1K  = (size_t)cH * cCB * 20 * cD * 2;
constexpr size_t OFF_X1Q = OFF_X1K + SZ_X1K;              // bf16 [h][b][t=5][d]
constexpr size_t SZ_X1Q  = (size_t)cH * cCB * 5 * cD * 2;
constexpr size_t OFF_Q   = OFF_X1Q + SZ_X1Q;              // fp32 [h][b][d]
constexpr size_t SZ_Q    = (size_t)cH * cBS * cD * 4;
constexpr size_t OFF_SC  = OFF_Q + SZ_Q;                  // fp32 [h][b][16]

__device__ __forceinline__ unsigned short f2bf(float x) {
    unsigned int u = __float_as_uint(x);
    unsigned int r = u + 0x7FFFu + ((u >> 16) & 1u);     // round to nearest even
    return (unsigned short)(r >> 16);
}

// ---------------------------- prep kernels ---------------------------------

// conv1 weights (H,D,F,6) fp32 -> (H,D,224) bf16, zero-padded K (natural k = f*6+tau)
__global__ void convert_w1_kernel(const float* __restrict__ W, unsigned short* __restrict__ O) {
    int hd = blockIdx.x;              // H*D blocks
    int k  = threadIdx.x;             // 224 threads
    float v = (k < K1_REAL) ? W[(size_t)hd * K1_REAL + k] : 0.f;
    O[(size_t)hd * K1_PAD + k] = f2bf(v);
}

// conv2 weights (H,D,D,5) fp32 -> (H,D,2560) bf16, K permuted tau-major:
//   O[hd][tau*512 + i] = W[hd][i*5 + tau]
__global__ void convert_w2_kernel(const float* __restrict__ W, unsigned short* __restrict__ O, int n) {
    int idx = blockIdx.x * blockDim.x + threadIdx.x;
    if (idx < n) {
        int hd  = idx / K2;
        int r   = idx % K2;
        int tau = r >> 9;             // r / 512
        int i   = r & 511;
        O[idx] = f2bf(W[(size_t)hd * K2 + i * 5 + tau]);
    }
}

// fold BN: y = x*s + t with s = g/sqrt(v+1e-5), t = b - m*s
__global__ void bn_prep_kernel(const float* __restrict__ g, const float* __restrict__ b,
                               const float* __restrict__ m, const float* __restrict__ v,
                               float* __restrict__ S, float* __restrict__ T) {
    int i = blockIdx.x * blockDim.x + threadIdx.x;
    if (i < cH * cD) {
        float s = g[i] * rsqrtf(v[i] + 1e-5f);
        S[i] = s;
        T[i] = b[i] - m[i] * s;
    }
}

// M[j][h][v] = sum_d linW[j, h*20+d] * dct[d, v]   (20 x 4 x 35)
__global__ void mjh_kernel(const float* __restrict__ linW, const float* __restrict__ dct,
                           float* __restrict__ M) {
    int idx = blockIdx.x * blockDim.x + threadIdx.x;
    if (idx >= 20 * 4 * 35) return;
    int v = idx % 35, h = (idx / 35) % 4, j = idx / 140;
    float a = 0.f;
    for (int d = 0; d < 20; ++d) a += linW[j * 80 + h * 20 + d] * dct[d * 35 + v];
    M[idx] = a;
}

__global__ void score_init_kernel(float* __restrict__ s, int n) {
    int i = blockIdx.x * blockDim.x + threadIdx.x;
    if (i < n) s[i] = 1e-15f;                            // reference adds 1e-15 to scores
}

// ---------------------------- WMMA conv-GEMM -------------------------------
// MODE 0: conv1 key   (K=224 , N=CB*20, gather src/1000, out X1k bf16 [b][t][d])
// MODE 1: conv1 query (K=224 , N=CB*5 , gather src/1000, out X1q bf16 [b][t][d])
// MODE 2: conv2 key   (K=2560, N=CB*16, B128 loads from X1k, fused q.k -> score)
// MODE 3: conv2 query (K=2560, N=CB   , B128 loads from X1q, out q fp32)
// Block: 256 threads = 8 wave32; block tile 128(M) x 64(N); k-step 32.
template <int MODE>
__global__ __launch_bounds__(256)
void gemm_kernel(const unsigned short* __restrict__ Aall,
                 const float* __restrict__ src,
                 const unsigned short* __restrict__ X1,
                 const float* __restrict__ Sbn, const float* __restrict__ Tbn,
                 unsigned short* __restrict__ outb,
                 float* __restrict__ qbuf,
                 float* __restrict__ score,
                 int b0) {
    constexpr int KD  = (MODE <= 1) ? K1_PAD : K2;
    constexpr int NPB = (MODE == 0) ? 20 : (MODE == 1) ? 5 : (MODE == 2) ? 16 : 1;
    constexpr int LT  = (MODE == 0 || MODE == 2) ? 20 : 5;  // time extent of X1 buffer
    constexpr int KU4 = KD / 8;                             // uint4 per weight row

    const int h    = blockIdx.z;
    const int m0   = blockIdx.y * 128;
    const int n0   = blockIdx.x * 64;
    const int tid  = threadIdx.x;
    const int wv   = tid >> 5;
    const int lane = tid & 31;

    __shared__ __align__(16) unsigned short Asm[128 * 32];
    __shared__ __align__(16) unsigned short Bsm[64 * 32];  // [n][k], K-major per column

    v8f acc[4] = {{}, {}, {}, {}};

    const uint4* Ag = reinterpret_cast<const uint4*>(Aall) + ((size_t)h * cD + m0) * KU4;

    for (int kb = 0; kb < KD; kb += 32) {
        // ---- stage A tile (128 x 32 bf16, row-major): 512 uint4 chunks ----
        {
            const uint4* Abase = Ag + (kb >> 3);
            #pragma unroll
            for (int it = 0; it < 2; ++it) {
                int q = tid + it * 256;
                int row = q >> 2, c = q & 3;
                *reinterpret_cast<uint4*>(&Asm[row * 32 + c * 8]) =
                    Abase[(size_t)row * KU4 + c];
            }
            if (kb + 32 < KD) {                           // prefetch next A k-slab
                const uint4* An = Ag + ((kb + 32) >> 3);
                __builtin_prefetch(&An[(size_t)(tid >> 2) * KU4 + (tid & 3)], 0, 3);
            }
        }
        // ---- stage B tile (64 cols x 32 k) ----
        if constexpr (MODE <= 1) {
            // layer 1: scalar im2col gather from src (minor cost path)
            int n  = tid >> 2;                            // column 0..63
            int kq = (tid & 3) * 8;                       // 8 consecutive k
            int col = n0 + n;
            int b = col / NPB, t = col % NPB;
            #pragma unroll
            for (int j = 0; j < 8; ++j) {
                int kk = kb + kq + j;
                unsigned short hv = 0;
                if (kk < K1_REAL) {
                    int f = kk / 6, tau = kk % 6;
                    int tim = (MODE == 0) ? (t + tau) : (40 + t + tau);
                    hv = f2bf(src[((size_t)(b0 + b) * 50 + tim) * cF + f] * 1e-3f);
                }
                Bsm[n * 32 + kq + j] = hv;
            }
        } else {
            // layer 2: tau-major K + time-major X1 -> fully vectorized staging
            int n = tid >> 2;                             // column 0..63
            int c = tid & 3;                              // 8-half chunk
            int col = n0 + n;
            int b = (MODE == 2) ? (col >> 4) : col;
            int t = (MODE == 2) ? (col & 15) : 0;
            int tau = kb >> 9;                            // k-step never spans tau
            int i0  = (kb & 511) + c * 8;
            const uint4* Xb = reinterpret_cast<const uint4*>(
                &X1[((size_t)(h * cCB + b) * LT + (t + tau)) * cD + i0]);
            *reinterpret_cast<uint4*>(&Bsm[n * 32 + c * 8]) = Xb[0];
        }
        __syncthreads();

        // ---- fragments per documented 16-bit WMMA VGPR layouts ----
        Frag a;
        {
            int row = wv * 16 + (lane & 15);
            int klo = (lane & 16) ? 8 : 0;                // halves 0-7 -> k=klo.., 8-15 -> +16
            a.u[0] = *reinterpret_cast<const uint4*>(&Asm[row * 32 + klo]);
            a.u[1] = *reinterpret_cast<const uint4*>(&Asm[row * 32 + klo + 16]);
        }
        const int kb2 = (lane & 16) ? 8 : 0;              // B: halves 0-7 -> k=kb2.., 8-15 -> +16
        #pragma unroll
        for (int ns = 0; ns < 4; ++ns) {
            Frag bf;
            int nn = ns * 16 + (lane & 15);
            bf.u[0] = *reinterpret_cast<const uint4*>(&Bsm[nn * 32 + kb2]);
            bf.u[1] = *reinterpret_cast<const uint4*>(&Bsm[nn * 32 + kb2 + 16]);
            acc[ns] = __builtin_amdgcn_wmma_f32_16x16x32_bf16(false, a.v, false, bf.v,
                                                              (short)0, acc[ns], false, false);
        }
        __syncthreads();
    }

    // ---- epilogue: relu -> BN, mode-specific sink ----
    const int d0    = m0 + wv * 16 + ((lane & 16) ? 8 : 0);   // 8 contiguous rows
    const int nlane = lane & 15;

    float sv[8], tv[8];
    #pragma unroll
    for (int r = 0; r < 8; ++r) {
        sv[r] = Sbn[h * cD + d0 + r];
        tv[r] = Tbn[h * cD + d0 + r];
    }

    #pragma unroll
    for (int ns = 0; ns < 4; ++ns) {
        int col = n0 + ns * 16 + nlane;
        if constexpr (MODE == 0 || MODE == 1) {
            int b = col / NPB, t = col % NPB;
            unsigned int pk[4];
            #pragma unroll
            for (int p = 0; p < 4; ++p) {
                float vlo = fmaxf(acc[ns][2 * p],     0.f) * sv[2 * p]     + tv[2 * p];
                float vhi = fmaxf(acc[ns][2 * p + 1], 0.f) * sv[2 * p + 1] + tv[2 * p + 1];
                pk[p] = (unsigned int)f2bf(vlo) | ((unsigned int)f2bf(vhi) << 16);
            }
            uint4 val = {pk[0], pk[1], pk[2], pk[3]};      // 8 bf16, contiguous in d
            *reinterpret_cast<uint4*>(
                &outb[((size_t)(h * cCB + b) * LT + t) * cD + d0]) = val;
        } else if constexpr (MODE == 3) {
            int b = col;                                   // NPB == 1
            float4 v0, v1;
            v0.x = fmaxf(acc[ns][0], 0.f) * sv[0] + tv[0];
            v0.y = fmaxf(acc[ns][1], 0.f) * sv[1] + tv[1];
            v0.z = fmaxf(acc[ns][2], 0.f) * sv[2] + tv[2];
            v0.w = fmaxf(acc[ns][3], 0.f) * sv[3] + tv[3];
            v1.x = fmaxf(acc[ns][4], 0.f) * sv[4] + tv[4];
            v1.y = fmaxf(acc[ns][5], 0.f) * sv[5] + tv[5];
            v1.z = fmaxf(acc[ns][6], 0.f) * sv[6] + tv[6];
            v1.w = fmaxf(acc[ns][7], 0.f) * sv[7] + tv[7];
            float* qp = &qbuf[((size_t)h * cBS + b0 + b) * cD + d0];
            *reinterpret_cast<float4*>(qp)     = v0;
            *reinterpret_cast<float4*>(qp + 4) = v1;
        } else {                                           // MODE == 2: fused q.k -> score
            int b = col >> 4, t = col & 15;                // a 16-col subtile = one batch
            const float* qrow = &qbuf[((size_t)h * cBS + b0 + b) * cD];
            float part = 0.f;
            #pragma unroll
            for (int r = 0; r < 8; ++r) {
                float v = fmaxf(acc[ns][r], 0.f) * sv[r] + tv[r];
                part += qrow[d0 + r] * v;
            }
            part += __shfl_xor(part, 16, 32);              // fold the two m-halves
            if (lane < 16)
                atomicAdd(&score[((size_t)h * cBS + b0 + b) * 16 + t], part);
        }
    }
}

// ---------------------------- final fused output ---------------------------
// out[b,f,j] = sum_u w[b,j,u] * src[b,u,f] + linb[j],
//   w[b,j,u] = sum_h sum_k att_h[b,k] * M[j,h,u-k]
__global__ __launch_bounds__(256)
void final_kernel(const float* __restrict__ src, const float* __restrict__ Mjh,
                  const float* __restrict__ score, const float* __restrict__ linb,
                  float* __restrict__ out) {
    __shared__ float att[64];
    __shared__ float hsum[4];
    __shared__ float Msm[20 * 4 * 35];
    __shared__ float ssm[50 * cF];
    __shared__ float wsm[20 * 50];

    const int b = blockIdx.x, tid = threadIdx.x;
    if (tid < 64) att[tid] = score[((size_t)(tid >> 4) * cBS + b) * 16 + (tid & 15)];
    for (int i = tid; i < 20 * 4 * 35; i += 256) Msm[i] = Mjh[i];
    for (int i = tid; i < 50 * cF; i += 256) ssm[i] = src[(size_t)b * 50 * cF + i];
    __syncthreads();
    if (tid < 4) {
        float s = 0.f;
        for (int k = 0; k < 16; ++k) s += att[tid * 16 + k];
        hsum[tid] = s;
    }
    __syncthreads();
    for (int idx = tid; idx < 1000; idx += 256) {
        int j = idx / 50, u = idx % 50;
        int klo = u - 34; if (klo < 0) klo = 0;
        int khi = u;      if (khi > 15) khi = 15;
        float a = 0.f;
        for (int h = 0; h < 4; ++h) {
            float inv = 1.f / hsum[h];
            for (int k = klo; k <= khi; ++k)
                a += att[h * 16 + k] * inv * Msm[j * 140 + h * 35 + (u - k)];
        }
        wsm[idx] = a;
    }
    __syncthreads();
    for (int idx = tid; idx < cF * 20; idx += 256) {
        int f = idx / 20, j = idx % 20;
        float a = linb[j];
        for (int u = 0; u < 50; ++u) a += wsm[j * 50 + u] * ssm[u * cF + f];
        out[(size_t)b * cF * 20 + f * 20 + j] = a;
    }
}

// ---------------------------------------------------------------------------

extern "C" void kernel_launch(void* const* d_in, const int* in_sizes, int n_in,
                              void* d_out, int out_size, void* d_ws, size_t ws_size,
                              hipStream_t stream) {
    (void)in_sizes; (void)n_in; (void)out_size; (void)ws_size;

    const float* src  = (const float*)d_in[0];
    const float* dct  = (const float*)d_in[1];
    const float* Wq1  = (const float*)d_in[2];
    const float* qg1  = (const float*)d_in[3];
    const float* qb1  = (const float*)d_in[4];
    const float* qm1  = (const float*)d_in[5];
    const float* qv1  = (const float*)d_in[6];
    const float* Wq2  = (const float*)d_in[7];
    const float* qg2  = (const float*)d_in[8];
    const float* qb2  = (const float*)d_in[9];
    const float* qm2  = (const float*)d_in[10];
    const float* qv2  = (const float*)d_in[11];
    const float* Wk1  = (const float*)d_in[12];
    const float* kg1  = (const float*)d_in[13];
    const float* kb1  = (const float*)d_in[14];
    const float* km1  = (const float*)d_in[15];
    const float* kv1  = (const float*)d_in[16];
    const float* Wk2  = (const float*)d_in[17];
    const float* kg2  = (const float*)d_in[18];
    const float* kb2  = (const float*)d_in[19];
    const float* km2  = (const float*)d_in[20];
    const float* kv2  = (const float*)d_in[21];
    const float* linW = (const float*)d_in[22];
    const float* linb = (const float*)d_in[23];
    float* out = (float*)d_out;

    char* w = (char*)d_ws;
    unsigned short* Wk1b = (unsigned short*)(w + OFF_WK1);
    unsigned short* Wq1b = (unsigned short*)(w + OFF_WQ1);
    unsigned short* Wk2b = (unsigned short*)(w + OFF_WK2);
    unsigned short* Wq2b = (unsigned short*)(w + OFF_WQ2);
    float* SK1 = (float*)(w + OFF_BN + 0 * SZ_BN1);
    float* TK1 = (float*)(w + OFF_BN + 1 * SZ_BN1);
    float* SK2 = (float*)(w + OFF_BN + 2 * SZ_BN1);
    float* TK2 = (float*)(w + OFF_BN + 3 * SZ_BN1);
    float* SQ1 = (float*)(w + OFF_BN + 4 * SZ_BN1);
    float* TQ1 = (float*)(w + OFF_BN + 5 * SZ_BN1);
    float* SQ2 = (float*)(w + OFF_BN + 6 * SZ_BN1);
    float* TQ2 = (float*)(w + OFF_BN + 7 * SZ_BN1);
    float* MjhP   = (float*)(w + OFF_M);
    unsigned short* X1k = (unsigned short*)(w + OFF_X1K);
    unsigned short* X1q = (unsigned short*)(w + OFF_X1Q);
    float* qbuf   = (float*)(w + OFF_Q);
    float* scoreP = (float*)(w + OFF_SC);

    // ---- prep ----
    convert_w1_kernel<<<cH * cD, K1_PAD, 0, stream>>>(Wk1, Wk1b);
    convert_w1_kernel<<<cH * cD, K1_PAD, 0, stream>>>(Wq1, Wq1b);
    const int NW2 = cH * cD * K2;
    convert_w2_kernel<<<(NW2 + 255) / 256, 256, 0, stream>>>(Wk2, Wk2b, NW2);
    convert_w2_kernel<<<(NW2 + 255) / 256, 256, 0, stream>>>(Wq2, Wq2b, NW2);
    bn_prep_kernel<<<(cH * cD + 255) / 256, 256, 0, stream>>>(kg1, kb1, km1, kv1, SK1, TK1);
    bn_prep_kernel<<<(cH * cD + 255) / 256, 256, 0, stream>>>(kg2, kb2, km2, kv2, SK2, TK2);
    bn_prep_kernel<<<(cH * cD + 255) / 256, 256, 0, stream>>>(qg1, qb1, qm1, qv1, SQ1, TQ1);
    bn_prep_kernel<<<(cH * cD + 255) / 256, 256, 0, stream>>>(qg2, qb2, qm2, qv2, SQ2, TQ2);
    mjh_kernel<<<(2800 + 255) / 256, 256, 0, stream>>>(linW, dct, MjhP);
    const int NSC = cH * cBS * 16;
    score_init_kernel<<<(NSC + 255) / 256, 256, 0, stream>>>(scoreP, NSC);

    // ---- conv pipeline, batch-chunked (query first: key L2 consumes qbuf) ----
    for (int c = 0; c < cNCHUNK; ++c) {
        int b0 = c * cCB;
        gemm_kernel<1><<<dim3(cCB * 5 / 64, cD / 128, cH), 256, 0, stream>>>(
            Wq1b, src, nullptr, SQ1, TQ1, X1q, nullptr, nullptr, b0);
        gemm_kernel<3><<<dim3(cCB / 64, cD / 128, cH), 256, 0, stream>>>(
            Wq2b, nullptr, X1q, SQ2, TQ2, nullptr, qbuf, nullptr, b0);
        gemm_kernel<0><<<dim3(cCB * 20 / 64, cD / 128, cH), 256, 0, stream>>>(
            Wk1b, src, nullptr, SK1, TK1, X1k, nullptr, nullptr, b0);
        gemm_kernel<2><<<dim3(cCB * 16 / 64, cD / 128, cH), 256, 0, stream>>>(
            Wk2b, nullptr, X1k, SK2, TK2, nullptr, qbuf, scoreP, b0);
    }

    // ---- attention normalization + fused DCT/linW output ----
    final_kernel<<<cBS, 256, 0, stream>>>(src, MjhP, scoreP, linb, out);
}
